// AoAReader_68272800137579
// MI455X (gfx1250) — matrix operand order repaired and checked by reading, not compiled, not run on hardware
//
#include <hip/hip_runtime.h>

// ---------------------------------------------------------------------------
// AoA Reader for MI455X (gfx1250, wave32, WMMA bf16 16x16x32)
//
// Sizes: NTOKENS=50000, E=256, H=256, NLAYERS=2, TD=2000, TQ=60, B=32, NC=10.
//
//  - All large GEMMs via v_wmma_f32_16x16x32_bf16 (f32 accumulate).
//  - gi = x @ Wih^T + bih precomputed as a parallel GEMM (bf16 out, bias fused).
//  - Recurrent scan: 1 workgroup/direction, 16 waves; h (32x256) in LDS
//    (f32 + bf16 mirror); 48 WMMAs/wave/step. Whh fragments for k<128 are
//    register-resident across the whole time loop (96 VGPRs); fragments for
//    k>=128 are streamed from L2 each step through a pointer made opaque with
//    an empty asm (blocks LICM -> no spills, but loads stay plain
//    global_load_b128 with batched waits). Next step's gi slab is prefetched
//    (global_prefetch_b8, near scope) to hide latency on the serial chain.
//  - Attention M = h_doc . h_query^T as batched WMMA GEMM (TQ padded to 64).
//  - Masked softmax stats / AoA combine / candidate sum: small VALU kernels.
// ---------------------------------------------------------------------------

typedef __bf16 bf16;
typedef __attribute__((ext_vector_type(16))) __bf16 v16bf;
typedef __attribute__((ext_vector_type(8)))  __bf16 v8bf;
typedef __attribute__((ext_vector_type(8)))  float  v8f;

#define TDc 2000
#define TQc 60
#define TQP 64      // padded query length for WMMA N-tiling
#define Bc  32
#define Ec  256
#define Hc  256
#define G3  768     // 3*H
#define NCc 10
#define EPSc 1e-12f

static __device__ __forceinline__ v8f vzero8() {
  v8f v = {0.f,0.f,0.f,0.f,0.f,0.f,0.f,0.f};
  return v;
}

// ---------------------------------------------------------------------------
// f32 -> bf16 conversion (weights)
// ---------------------------------------------------------------------------
__global__ void k_f32_to_bf16(const float* __restrict__ s, bf16* __restrict__ d, int n) {
  int i = blockIdx.x * blockDim.x + threadIdx.x;
  if (i < n) d[i] = (bf16)s[i];
}

// ---------------------------------------------------------------------------
// Embedding gather: out[row, :] = bf16(embed[tok[row], :]), row = t*B + b
// ---------------------------------------------------------------------------
__global__ void k_embed(const int* __restrict__ tok, const float* __restrict__ emb,
                        bf16* __restrict__ out, int rows) {
  int i = blockIdx.x * blockDim.x + threadIdx.x;
  if (i >= rows * (Ec / 8)) return;
  int row = i >> 5;          // Ec/8 == 32
  int sub = i & 31;
  int t = tok[row];
  const float* s = emb + (long)t * Ec + sub * 8;
  v8bf v;
#pragma unroll
  for (int j = 0; j < 8; ++j) v[j] = (bf16)s[j];
  *(v8bf*)(out + (long)row * Ec + sub * 8) = v;
}

// ---------------------------------------------------------------------------
// Batched WMMA bf16 GEMM:  C[z][M,N] = A[z][M,K] * B[z][N,K]^T (+bias)
// 8 waves (4 in M x 2 in N); wave tile = (MT*16) x (NT*16).
// Block tile = 128 x (NT*32). A/B fragments use contiguous k = 16*hi + j in both; C per ISA (wave32).
// ---------------------------------------------------------------------------
template<int MT, int NT, bool BF16OUT, bool BIAS, bool MGUARD>
__global__ void __launch_bounds__(256)
k_gemm(const bf16* __restrict__ A, long aBatch, int lda,
       const bf16* __restrict__ Bw, long bBatch, int ldb,
       void* __restrict__ Cout, long cBatch, int ldc,
       const float* __restrict__ bias, int M, int N, int K)
{
  int z = blockIdx.z;
  const bf16* Ab = A + (long)z * aBatch;
  const bf16* Bb = Bw + (long)z * bBatch;
  int tid = threadIdx.x, lane = tid & 31, w = tid >> 5;
  int wm = w & 3, wn = w >> 2;
  int m0 = blockIdx.x * (4 * MT * 16) + wm * (MT * 16);
  int n0 = blockIdx.y * (2 * NT * 16) + wn * (NT * 16);
  int lo = lane & 15, hi = lane >> 4;

  v8f acc[MT][NT];
#pragma unroll
  for (int i = 0; i < MT; ++i)
#pragma unroll
    for (int j = 0; j < NT; ++j) acc[i][j] = vzero8();

  for (int k0 = 0; k0 < K; k0 += 32) {
    v16bf af[MT], bfrag[NT];
#pragma unroll
    for (int mt = 0; mt < MT; ++mt) {
      int row = m0 + mt * 16 + lo;
      if (MGUARD) { if (row >= M) row = 0; }        // clamp; masked at store
      const bf16* p = Ab + (long)row * lda + k0 + 16 * hi;
      v8bf c0 = *(const v8bf*)p;
      v8bf c1 = *(const v8bf*)(p + 8);
#pragma unroll
      for (int j = 0; j < 8; ++j) { af[mt][j] = c0[j]; af[mt][8 + j] = c1[j]; }
    }
#pragma unroll
    for (int nt = 0; nt < NT; ++nt) {
      const bf16* p = Bb + (long)(n0 + nt * 16 + lo) * ldb + k0 + 16 * hi;
      v8bf c0 = *(const v8bf*)p;
      v8bf c1 = *(const v8bf*)(p + 8);
#pragma unroll
      for (int j = 0; j < 8; ++j) { bfrag[nt][j] = c0[j]; bfrag[nt][8 + j] = c1[j]; }
    }
#pragma unroll
    for (int mt = 0; mt < MT; ++mt)
#pragma unroll
      for (int nt = 0; nt < NT; ++nt)
        acc[mt][nt] = __builtin_amdgcn_wmma_f32_16x16x32_bf16(
            false, af[mt], false, bfrag[nt], (short)0, acc[mt][nt], false, false);
  }

#pragma unroll
  for (int mt = 0; mt < MT; ++mt)
#pragma unroll
    for (int nt = 0; nt < NT; ++nt) {
      int col = n0 + nt * 16 + lo;
      float bv = BIAS ? bias[col] : 0.f;
#pragma unroll
      for (int r = 0; r < 8; ++r) {
        int row = m0 + mt * 16 + r + 8 * hi;        // C layout: vgpr r -> M=r(+8)
        if (!MGUARD || row < M) {
          float v = acc[mt][nt][r] + bv;
          long off = (long)z * cBatch + (long)row * ldc + col;
          if (BF16OUT) ((bf16*)Cout)[off] = (bf16)v;
          else         ((float*)Cout)[off] = v;
        }
      }
    }
}

// ---------------------------------------------------------------------------
// Bidirectional GRU scan for one layer. grid.x = 2 (dir 0 fwd, dir 1 bwd).
// 512 threads = 16 waves; wave w owns output features [w*16, w*16+16).
// ---------------------------------------------------------------------------
#define KRES 128   // k-range whose Whh fragments stay register-resident

__global__ void __launch_bounds__(512)
k_gru_scan(const bf16* __restrict__ gi, long giDirStride,
           const bf16* __restrict__ Whh,            // [2][768][256] bf16
           const float* __restrict__ bhh_f, const float* __restrict__ bhh_b,
           bf16* __restrict__ Y, int T)
{
  __shared__ float hs[Bc * Hc];      // 32 KB, f32 hidden state
  __shared__ bf16  hb[Bc * Hc];      // 16 KB, bf16 mirror for WMMA A
  int dir = blockIdx.x;
  const bf16*  Wd  = Whh + (long)dir * G3 * Hc;
  const float* bhh = dir ? bhh_b : bhh_f;
  const bf16*  gid = gi + (long)dir * giDirStride;

  int tid = threadIdx.x, lane = tid & 31, w = tid >> 5;
  int lo = lane & 15, hi = lane >> 4;
  int nf = w * 16 + lo;              // this lane's output feature (0..255)
  float br = bhh[nf], bz = bhh[Hc + nf], bn = bhh[2 * Hc + nf];

  // Register-resident Whh fragments for k0 in [0, KRES): 12 x v16bf = 96 VGPRs,
  // loaded once and live across the whole time loop.
  v16bf bres[KRES / 32][3];
#pragma unroll
  for (int kk = 0; kk < KRES / 32; ++kk)
#pragma unroll
    for (int g = 0; g < 3; ++g) {
      int n = g * Hc + w * 16 + lo;
      const bf16* p = Wd + (long)n * Hc + kk * 32 + 16 * hi;
      v8bf c0 = *(const v8bf*)p;
      v8bf c1 = *(const v8bf*)(p + 8);
#pragma unroll
      for (int j = 0; j < 8; ++j) { bres[kk][g][j] = c0[j]; bres[kk][g][8 + j] = c1[j]; }
    }

  for (int i = tid; i < Bc * Hc; i += 512) { hs[i] = 0.f; hb[i] = (bf16)0.f; }
  __syncthreads();

  for (int tt = 0; tt < T; ++tt) {
    int t = dir ? (T - 1 - tt) : tt;

    // prefetch next timestep's gi slab into near cache: Bc*G3 bf16 = 24576
    // elems; 512 threads x 96-elem (192 B) stride spans 49152 elems (two slabs).
    if (tt + 1 < T) {
      int tn = dir ? (T - 2 - tt) : (tt + 1);
      __builtin_prefetch(gid + ((long)tn * Bc) * G3 + (long)tid * 96, 0, 3);
    }

    // Opaque zero offset: re-produced by asm every iteration so the streamed
    // Whh loads below are loop-variant (no LICM/hoist-and-spill), yet remain
    // plain global_load_b128 with normal scope and batched waits.
    int zoff = 0;
    asm volatile("" : "+s"(zoff));
    const bf16* Wdyn = Wd + zoff;

    v8f acc[2][3];
#pragma unroll
    for (int i = 0; i < 2; ++i)
#pragma unroll
      for (int g = 0; g < 3; ++g) acc[i][g] = vzero8();

#pragma unroll
    for (int k0 = 0; k0 < Hc; k0 += 32) {
      v16bf af[2];
#pragma unroll
      for (int mt = 0; mt < 2; ++mt) {
        int idx = (mt * 16 + lo) * Hc + k0 + 16 * hi;
        v8bf c0 = *(const v8bf*)&hb[idx];
        v8bf c1 = *(const v8bf*)&hb[idx + 8];
#pragma unroll
        for (int j = 0; j < 8; ++j) { af[mt][j] = c0[j]; af[mt][8 + j] = c1[j]; }
      }
      v16bf bfrag[3];
      if (k0 < KRES) {
#pragma unroll
        for (int g = 0; g < 3; ++g) bfrag[g] = bres[k0 / 32][g];
      } else {
        // streamed from L2 every step (addresses depend on opaque zoff)
#pragma unroll
        for (int g = 0; g < 3; ++g) {
          int n = g * Hc + w * 16 + lo;
          const bf16* p = Wdyn + (long)n * Hc + k0 + 16 * hi;
          v8bf c0 = *(const v8bf*)p;
          v8bf c1 = *(const v8bf*)(p + 8);
#pragma unroll
          for (int j = 0; j < 8; ++j) { bfrag[g][j] = c0[j]; bfrag[g][8 + j] = c1[j]; }
        }
      }
#pragma unroll
      for (int mt = 0; mt < 2; ++mt)
#pragma unroll
        for (int g = 0; g < 3; ++g)
          acc[mt][g] = __builtin_amdgcn_wmma_f32_16x16x32_bf16(
              false, af[mt], false, bfrag[g], (short)0, acc[mt][g], false, false);
    }
    __syncthreads();   // all reads of hb complete

#pragma unroll
    for (int mt = 0; mt < 2; ++mt) {
#pragma unroll
      for (int r = 0; r < 8; ++r) {
        int bi = mt * 16 + r + 8 * hi;                 // batch index (C layout)
        long grow = ((long)t * Bc + bi) * G3;
        float ir  = (float)gid[grow + nf];
        float iz  = (float)gid[grow + Hc + nf];
        float in_ = (float)gid[grow + 2 * Hc + nf];
        float hr = acc[mt][0][r] + br;
        float hz = acc[mt][1][r] + bz;
        float hn = acc[mt][2][r] + bn;
        float hp = hs[bi * Hc + nf];
        float rg = 1.f / (1.f + __expf(-(ir + hr)));
        float zg = 1.f / (1.f + __expf(-(iz + hz)));
        float ng = tanhf(in_ + rg * hn);
        float hnw = (1.f - zg) * ng + zg * hp;
        hs[bi * Hc + nf] = hnw;
        hb[bi * Hc + nf] = (bf16)hnw;
        Y[((long)t * Bc + bi) * (2 * Hc) + dir * Hc + nf] = (bf16)hnw;
      }
    }
    __syncthreads();   // writes visible before next step's WMMA reads
  }
}

// ---------------------------------------------------------------------------
// Per (b,d): rowmax = max_q M, rowsum = dmask * sum_q exp(M-rowmax)*qmask
// ---------------------------------------------------------------------------
__global__ void k_row_stats(const float* __restrict__ Mb, const float* __restrict__ dmask,
                            const float* __restrict__ qmask,
                            float* __restrict__ rmax, float* __restrict__ rsum) {
  int idx = blockIdx.x * blockDim.x + threadIdx.x;   // b*TD + d
  if (idx >= Bc * TDc) return;
  int b = idx / TDc, d = idx % TDc;
  const float* row = Mb + (long)idx * TQP;
  float mx = -3.4e38f;
  for (int q = 0; q < TQc; ++q) mx = fmaxf(mx, row[q]);
  float s = 0.f;
  for (int q = 0; q < TQc; ++q) s += __expf(row[q] - mx) * qmask[q * Bc + b];
  rmax[idx] = mx;
  rsum[idx] = s * dmask[d * Bc + b];
}

// ---------------------------------------------------------------------------
// Per (b,q): colmax/colsum over d, plus average_beta. grid = (TQ, B).
// ---------------------------------------------------------------------------
__global__ void __launch_bounds__(256)
k_col_stats(const float* __restrict__ Mb, const float* __restrict__ dmask,
            const float* __restrict__ qmask, const float* __restrict__ dlen,
            const float* __restrict__ rmax, const float* __restrict__ rsum,
            float* __restrict__ cmax, float* __restrict__ csum, float* __restrict__ abeta) {
  int q = blockIdx.x, b = blockIdx.y, tid = threadIdx.x;
  __shared__ float red[256], red2[256];
  float mx = -3.4e38f;
  for (int d = tid; d < TDc; d += 256) mx = fmaxf(mx, Mb[((long)b * TDc + d) * TQP + q]);
  red[tid] = mx; __syncthreads();
  for (int s = 128; s > 0; s >>= 1) { if (tid < s) red[tid] = fmaxf(red[tid], red[tid + s]); __syncthreads(); }
  mx = red[0];
  __syncthreads();
  float cs = 0.f, an = 0.f;
  for (int d = tid; d < TDc; d += 256) {
    float m = Mb[((long)b * TDc + d) * TQP + q];
    float dm = dmask[d * Bc + b];
    cs += __expf(m - mx) * dm;
    int ridx = b * TDc + d;
    an += __expf(m - rmax[ridx]) * dm / (rsum[ridx] + EPSc);
  }
  red[tid] = cs; red2[tid] = an; __syncthreads();
  for (int s = 128; s > 0; s >>= 1) {
    if (tid < s) { red[tid] += red[tid + s]; red2[tid] += red2[tid + s]; }
    __syncthreads();
  }
  if (tid == 0) {
    float qm = qmask[q * Bc + b];
    cmax[b * TQP + q]  = mx;
    csum[b * TQP + q]  = red[0] * qm;
    abeta[b * TQP + q] = red2[0] * qm / dlen[b];
  }
}

// ---------------------------------------------------------------------------
// Per b: s[d] = dmask * sum_q exp(M-colmax)*qmask/(colsum+eps)*avg_beta;
// probs[c] = sum over d where doc token == candidate token. grid = B.
// ---------------------------------------------------------------------------
__global__ void __launch_bounds__(256)
k_probs(const float* __restrict__ Mb, const int* __restrict__ docs,
        const float* __restrict__ dmask, const float* __restrict__ qmask,
        const int* __restrict__ cand, const float* __restrict__ cmax,
        const float* __restrict__ csum, const float* __restrict__ abeta,
        float* __restrict__ out) {
  int b = blockIdx.x, tid = threadIdx.x;
  __shared__ float wq[TQc], cm[TQc], cp[NCc];
  __shared__ int ct[NCc];
  if (tid < TQc) {
    float qm = qmask[tid * Bc + b];
    wq[tid] = qm / (csum[b * TQP + tid] + EPSc) * abeta[b * TQP + tid];
    cm[tid] = cmax[b * TQP + tid];
  }
  if (tid < NCc) { cp[tid] = 0.f; ct[tid] = cand[b * NCc + tid]; }
  __syncthreads();
  for (int d = tid; d < TDc; d += 256) {
    const float* row = Mb + ((long)b * TDc + d) * TQP;
    float sd = 0.f;
    for (int q = 0; q < TQc; ++q) sd += __expf(row[q] - cm[q]) * wq[q];
    sd *= dmask[d * Bc + b];
    int tk = docs[d * Bc + b];
    for (int c = 0; c < NCc; ++c)
      if (tk == ct[c]) atomicAdd(&cp[c], sd);
  }
  __syncthreads();
  if (tid < NCc) out[b * NCc + tid] = cp[tid];
}

// ---------------------------------------------------------------------------
// Host orchestration
// ---------------------------------------------------------------------------
extern "C" void kernel_launch(void* const* d_in, const int* in_sizes, int n_in,
                              void* d_out, int out_size, void* d_ws, size_t ws_size,
                              hipStream_t stream) {
  const int*   documents = (const int*)d_in[0];
  const float* dmask     = (const float*)d_in[1];
  const float* dlen      = (const float*)d_in[2];
  const int*   query     = (const int*)d_in[3];
  const float* qmask     = (const float*)d_in[4];
  const int*   cand      = (const int*)d_in[5];
  const float* embed     = (const float*)d_in[6];
  // gru params: [layer][dir][{Wih,Whh,bih,bhh}] starting at index 7
  auto P = [&](int layer, int dir, int which) -> const float* {
    return (const float*)d_in[7 + layer * 8 + dir * 4 + which];
  };

  char* ws = (char*)d_ws;
  size_t o = 0;
  auto take = [&](size_t bytes) { size_t r = o; o += (bytes + 255) & ~(size_t)255; return r; };

  const size_t WIH0 = (size_t)G3 * Ec;        // 196608
  const size_t WIH1 = (size_t)G3 * (2 * Hc);  // 393216
  const size_t WHH  = (size_t)G3 * Hc;        // 196608

  size_t oWih0f = take(WIH0 * 2), oWih0b = take(WIH0 * 2);
  size_t oWhh0  = take(2 * WHH * 2);          // [2][768][256]
  size_t oWih1f = take(WIH1 * 2), oWih1b = take(WIH1 * 2);
  size_t oWhh1  = take(2 * WHH * 2);
  size_t oXd0   = take((size_t)TDc * Bc * Ec * 2);
  size_t oYd0   = take((size_t)TDc * Bc * 2 * Hc * 2);
  size_t oYd1   = take((size_t)TDc * Bc * 2 * Hc * 2);
  size_t oXq0   = take((size_t)TQP * Bc * Ec * 2);
  size_t oYq0   = take((size_t)TQP * Bc * 2 * Hc * 2);
  size_t oYq1   = take((size_t)TQP * Bc * 2 * Hc * 2);
  size_t oGI    = take((size_t)2 * TDc * Bc * G3 * 2);   // reused doc & query
  size_t oM     = take((size_t)Bc * TDc * TQP * 4);
  size_t oRmax  = take((size_t)Bc * TDc * 4);
  size_t oRsum  = take((size_t)Bc * TDc * 4);
  size_t oCmax  = take((size_t)Bc * TQP * 4);
  size_t oCsum  = take((size_t)Bc * TQP * 4);
  size_t oAbeta = take((size_t)Bc * TQP * 4);
  (void)ws_size; (void)n_in; (void)in_sizes; (void)out_size;

  bf16* Wih0[2] = {(bf16*)(ws + oWih0f), (bf16*)(ws + oWih0b)};
  bf16* Whh0    = (bf16*)(ws + oWhh0);
  bf16* Wih1[2] = {(bf16*)(ws + oWih1f), (bf16*)(ws + oWih1b)};
  bf16* Whh1    = (bf16*)(ws + oWhh1);
  bf16* Xd0 = (bf16*)(ws + oXd0);
  bf16* Yd0 = (bf16*)(ws + oYd0);
  bf16* Yd1 = (bf16*)(ws + oYd1);
  bf16* Xq0 = (bf16*)(ws + oXq0);
  bf16* Yq0 = (bf16*)(ws + oYq0);
  bf16* Yq1 = (bf16*)(ws + oYq1);
  bf16* GI  = (bf16*)(ws + oGI);
  float* Mbuf  = (float*)(ws + oM);
  float* Rmax  = (float*)(ws + oRmax);
  float* Rsum  = (float*)(ws + oRsum);
  float* Cmax  = (float*)(ws + oCmax);
  float* Csum  = (float*)(ws + oCsum);
  float* Abeta = (float*)(ws + oAbeta);

  auto cvt = [&](const float* src, bf16* dst, int n) {
    k_f32_to_bf16<<<(n + 255) / 256, 256, 0, stream>>>(src, dst, n);
  };
  for (int dir = 0; dir < 2; ++dir) {
    cvt(P(0, dir, 0), Wih0[dir], (int)WIH0);
    cvt(P(0, dir, 1), Whh0 + (size_t)dir * WHH, (int)WHH);
    cvt(P(1, dir, 0), Wih1[dir], (int)WIH1);
    cvt(P(1, dir, 1), Whh1 + (size_t)dir * WHH, (int)WHH);
  }

  // embedding gather
  k_embed<<<(TDc * Bc * (Ec / 8) + 255) / 256, 256, 0, stream>>>(documents, embed, Xd0, TDc * Bc);
  k_embed<<<(TQc * Bc * (Ec / 8) + 255) / 256, 256, 0, stream>>>(query, embed, Xq0, TQc * Bc);

  const long giStrideDoc = (long)TDc * Bc * G3;
  const long giStrideQ   = (long)TQc * Bc * G3;

  // gi GEMMs: block tile 128x128 (wave tile 32x64), no M-guard (M % 128 == 0)
  auto gi_gemm = [&](const bf16* X, int Kdim, bf16* Wih, const float* bih,
                     bf16* gi_out, int Mrows) {
    k_gemm<2, 4, true, true, false><<<dim3(Mrows / 128, G3 / 128, 1), 256, 0, stream>>>(
        X, 0, Kdim, Wih, 0, Kdim, gi_out, 0, G3, bih, Mrows, G3, Kdim);
  };

  // ---- document GRU ----
  for (int dir = 0; dir < 2; ++dir)
    gi_gemm(Xd0, Ec, Wih0[dir], P(0, dir, 2), GI + dir * giStrideDoc, TDc * Bc);
  k_gru_scan<<<2, 512, 0, stream>>>(GI, giStrideDoc, Whh0, P(0, 0, 3), P(0, 1, 3), Yd0, TDc);
  for (int dir = 0; dir < 2; ++dir)
    gi_gemm(Yd0, 2 * Hc, Wih1[dir], P(1, dir, 2), GI + dir * giStrideDoc, TDc * Bc);
  k_gru_scan<<<2, 512, 0, stream>>>(GI, giStrideDoc, Whh1, P(1, 0, 3), P(1, 1, 3), Yd1, TDc);

  // ---- query GRU ----
  for (int dir = 0; dir < 2; ++dir)
    gi_gemm(Xq0, Ec, Wih0[dir], P(0, dir, 2), GI + dir * giStrideQ, TQc * Bc);
  k_gru_scan<<<2, 512, 0, stream>>>(GI, giStrideQ, Whh0, P(0, 0, 3), P(0, 1, 3), Yq0, TQc);
  for (int dir = 0; dir < 2; ++dir)
    gi_gemm(Yq0, 2 * Hc, Wih1[dir], P(1, dir, 2), GI + dir * giStrideQ, TQc * Bc);
  k_gru_scan<<<2, 512, 0, stream>>>(GI, giStrideQ, Whh1, P(1, 0, 3), P(1, 1, 3), Yq1, TQc);

  // ---- attention M[b] = h_doc[b] (TD x 512) . h_query[b]^T (512 x TQP) ----
  k_gemm<2, 2, false, false, true><<<dim3((TDc + 127) / 128, TQP / 64, Bc), 256, 0, stream>>>(
      Yd1, 2 * Hc, Bc * 2 * Hc, Yq1, 2 * Hc, Bc * 2 * Hc,
      Mbuf, (long)TDc * TQP, TQP, nullptr, TDc, TQP, 2 * Hc);

  // ---- softmax stats + AoA combine + candidate sum ----
  k_row_stats<<<(Bc * TDc + 255) / 256, 256, 0, stream>>>(Mbuf, dmask, qmask, Rmax, Rsum);
  k_col_stats<<<dim3(TQc, Bc, 1), 256, 0, stream>>>(Mbuf, dmask, qmask, dlen, Rmax, Rsum,
                                                    Cmax, Csum, Abeta);
  k_probs<<<Bc, 256, 0, stream>>>(Mbuf, documents, dmask, qmask, cand,
                                  Cmax, Csum, Abeta, (float*)d_out);
}
